// SRNN_32040456028836
// MI455X (gfx1250) — matrix operand order repaired and checked by reading, not compiled, not run on hardware
//
#include <hip/hip_runtime.h>
#include <hip/hip_bf16.h>

// Problem dims
#define B_    16
#define T_    512
#define CENC_ 128
#define EMB_  512
#define H_    512
#define HH_   256
#define NL_   2
#define P_    8
#define NTAG_ 12
#define G4_   1024      // 4*HH
#define NR_   8192      // T*B rows

typedef __attribute__((ext_vector_type(16))) __bf16 v16bf;
typedef __attribute__((ext_vector_type(8)))  __bf16 v8bf;
typedef __attribute__((ext_vector_type(8)))  float  v8f;
typedef __bf16 bf16;

__device__ __forceinline__ float sigf(float x) { return 1.0f / (1.0f + __expf(-x)); }

// ---------------------------------------------------------------------------
// zero fill
__global__ void k_fill0(float* p, int n) {
    int i = blockIdx.x * blockDim.x + threadIdx.x;
    if (i < n) p[i] = 0.f;
}

// ---------------------------------------------------------------------------
// Generic transpose+convert: src [N][K] f32 -> dst [K][N] bf16
__global__ void k_t(const float* __restrict__ src, bf16* __restrict__ dst, int N, int K) {
    int i = blockIdx.x * blockDim.x + threadIdx.x;
    if (i >= N * K) return;
    int k = i / N, n = i - k * N;
    dst[i] = (bf16)src[(size_t)n * K + k];
}

// ---------------------------------------------------------------------------
// WMMA GEMM: C[M,N] = A[M,K] * WT[K,N] + bias1 + bias2
//   A: f32 row-major. WT: bf16, pre-transposed weights (row stride N).
// Block: 256 threads (8 waves), tile 128x128, wave tile 32x64, K stepped 32.
// REQUIRES: M%128==0, N%128==0, K%32==0 (all call sites satisfy this).
// A tile staged in LDS pre-swizzled into WMMA A-fragment order (each fragment
// is one contiguous 32B v16bf). B fragments load straight from global WT
// (16 bf16 loads per lane, row stride N; reused across all M-blocks -> L2 hot).
__global__ __launch_bounds__(256) void k_gemm(
    const float* __restrict__ A, int lda,
    const bf16* __restrict__ WT,
    const float* __restrict__ b1, const float* __restrict__ b2,
    float* __restrict__ C, int ldc, int M, int N, int K)
{
    __shared__ __attribute__((aligned(32))) bf16 As[8][32][16]; // [m-tile][lane][elem]

    const int tid = threadIdx.x;
    const int w = tid >> 5, L = tid & 31;
    const int m0 = blockIdx.y * 128, n0 = blockIdx.x * 128;
    const int wm = (w >> 1) * 32, wn = (w & 1) * 64;
    const int lm = tid >> 1;           // 0..127: tile row this thread stages
    const int kc = (tid & 1) * 16;     // k chunk base (0 or 16)

    v8f acc[2][4];
#pragma unroll
    for (int ms = 0; ms < 2; ++ms)
#pragma unroll
        for (int ns = 0; ns < 4; ++ns)
#pragma unroll
            for (int r = 0; r < 8; ++r) acc[ms][ns][r] = 0.f;

    const float* arow = A + (size_t)(m0 + lm) * lda + kc;
    // The 16 staged elements land in exactly two contiguous 8xbf16 chunks:
    //   k = kc..kc+7   -> lane (lm&15),    elems kc/2..kc/2+7
    //   k = kc+8..kc+15-> lane (lm&15)+16, elems kc/2..kc/2+7
    bf16* st0 = &As[lm >> 4][(lm & 15)][kc >> 1];
    bf16* st1 = &As[lm >> 4][(lm & 15) + 16][kc >> 1];

    for (int k0 = 0; k0 < K; k0 += 32) {
        float4 f0 = *(const float4*)(arow + k0);
        float4 f1 = *(const float4*)(arow + k0 + 4);
        float4 f2 = *(const float4*)(arow + k0 + 8);
        float4 f3 = *(const float4*)(arow + k0 + 12);
        v8bf c0, c1;
        c0[0] = (bf16)f0.x; c0[1] = (bf16)f0.y; c0[2] = (bf16)f0.z; c0[3] = (bf16)f0.w;
        c0[4] = (bf16)f1.x; c0[5] = (bf16)f1.y; c0[6] = (bf16)f1.z; c0[7] = (bf16)f1.w;
        c1[0] = (bf16)f2.x; c1[1] = (bf16)f2.y; c1[2] = (bf16)f2.z; c1[3] = (bf16)f2.w;
        c1[4] = (bf16)f3.x; c1[5] = (bf16)f3.y; c1[6] = (bf16)f3.z; c1[7] = (bf16)f3.w;
        *(v8bf*)st0 = c0;
        *(v8bf*)st1 = c1;
        __syncthreads();

        v16bf af[2], bfr[4];
#pragma unroll
        for (int ms = 0; ms < 2; ++ms)
            af[ms] = *(const v16bf*)&As[(w >> 1) * 2 + ms][L][0];
        const bf16* wrow = WT + (size_t)k0 * N + n0 + wn + (L & 15);
#pragma unroll
        for (int ns = 0; ns < 4; ++ns)
#pragma unroll
            for (int s = 0; s < 16; ++s)
                bfr[ns][s] = wrow[(size_t)(((L >> 4) << 3) + (s & 7) + ((s >> 3) << 4)) * N + ns * 16];
#pragma unroll
        for (int ms = 0; ms < 2; ++ms)
#pragma unroll
            for (int ns = 0; ns < 4; ++ns)
                acc[ms][ns] = __builtin_amdgcn_wmma_f32_16x16x32_bf16(
                    false, af[ms], false, bfr[ns], (short)0, acc[ms][ns], false, false);
        __syncthreads();
    }

    // D fragment: lane holds column n = L&15, rows m = 8*(L>>4)+r
#pragma unroll
    for (int ms = 0; ms < 2; ++ms)
#pragma unroll
        for (int ns = 0; ns < 4; ++ns) {
            int n = n0 + wn + ns * 16 + (L & 15);
            float bias = (b1 ? b1[n] : 0.f) + (b2 ? b2[n] : 0.f);
#pragma unroll
            for (int r = 0; r < 8; ++r) {
                int m = m0 + wm + ms * 16 + ((L >> 4) << 3) + r;
                C[(size_t)m * ldc + n] = acc[ms][ns][r] + bias;
            }
        }
}

// ---------------------------------------------------------------------------
// Persistent encoder LSTM recurrence (one block per direction, 16 waves).
// Per step: preact = h @ Whh^T via WMMA (Whh pre-transposed bf16 in global),
// + precomputed xW (+biases folded there), gate nonlinearity, packed-seq mask.
// Wave w owns hh columns [16w,16w+16): its 4 WMMA n-tiles are the matching
// i/f/g/o gate columns (gate*256 + 16w), so the cell update is wave-local.
__global__ __launch_bounds__(512) void k_enc(
    const float* __restrict__ xWf, const float* __restrict__ xWr, // [8192][1024]
    const bf16* __restrict__ WtBase,                              // [4][256][1024]
    const float* __restrict__ h0, const float* __restrict__ c0,   // [4][16][256]
    const int* __restrict__ lens,
    float* __restrict__ outp,                                     // [8192][512]
    int layer)
{
    const int dir = blockIdx.x;
    const float* xW = dir ? xWr : xWf;
    const bf16* Wt = WtBase + (size_t)(2 * layer + dir) * HH_ * G4_;
    const float* h0p = h0 + (size_t)(2 * layer + dir) * B_ * HH_;
    const float* c0p = c0 + (size_t)(2 * layer + dir) * B_ * HH_;

    __shared__ __attribute__((aligned(32))) bf16 Hs[8][32][16]; // swizzled h (A-frags)
    __shared__ int lens_s[B_];

    const int tid = threadIdx.x;
    const int w = tid >> 5, L = tid & 31;
    const int hh = w * 16 + (L & 15);
    const int bbase = (L >> 4) << 3;

    if (tid < B_) lens_s[tid] = lens[tid];
    for (int e = tid; e < B_ * HH_; e += 512) {
        int b = e >> 8, k = e & 255;
        Hs[k >> 5][b + ((((k & 31) >> 3) & 1) << 4)][((k & 31) & 7) + (((k & 31) >> 4) << 3)] =
            (bf16)h0p[e];
    }
    float c[8];
#pragma unroll
    for (int r = 0; r < 8; ++r) c[r] = c0p[(size_t)(bbase + r) * HH_ + hh];
    __syncthreads();

    for (int t = 0; t < T_; ++t) {
        v16bf a[8];
#pragma unroll
        for (int ks = 0; ks < 8; ++ks) a[ks] = *(const v16bf*)&Hs[ks][L][0];
        __syncthreads(); // all Hs reads complete before this step's writes

        v8f acc[4];
#pragma unroll
        for (int j = 0; j < 4; ++j)
#pragma unroll
            for (int r = 0; r < 8; ++r) acc[j][r] = 0.f;

#pragma unroll
        for (int ks = 0; ks < 8; ++ks) {
#pragma unroll
            for (int j = 0; j < 4; ++j) {
                v16bf bf;
#pragma unroll
                for (int s = 0; s < 16; ++s)
                    bf[s] = Wt[(size_t)(ks * 32 + ((L >> 4) << 3) + (s & 7) + ((s >> 3) << 4)) * G4_ + j * 256 + w * 16 + (L & 15)];
                acc[j] = __builtin_amdgcn_wmma_f32_16x16x32_bf16(
                    false, a[ks], false, bf, (short)0, acc[j], false, false);
            }
        }

#pragma unroll
        for (int r = 0; r < 8; ++r) {
            int b = bbase + r;
            int lv = lens_s[b];
            int row;
            if (dir == 0) row = t * B_ + b;
            else { int p = lv - 1 - t; row = (p < 0 ? 0 : p) * B_ + b; }
            const float* xr = xW + (size_t)row * G4_;
            float gi = acc[0][r] + xr[hh];
            float gf = acc[1][r] + xr[256 + hh];
            float gg = acc[2][r] + xr[512 + hh];
            float go = acc[3][r] + xr[768 + hh];
            float c2 = sigf(gf) * c[r] + sigf(gi) * tanhf(gg);
            float h2 = sigf(go) * tanhf(c2);
            bool m = (t < lv);
            if (dir == 0) {
                outp[(size_t)(t * B_ + b) * H_ + hh] = m ? h2 : 0.f;
            } else if (m) {
                outp[(size_t)((lv - 1 - t) * B_ + b) * H_ + HH_ + hh] = h2;
            }
            if (m) {
                c[r] = c2;
                int kk = hh & 31;
                Hs[hh >> 5][b + (((kk >> 3) & 1) << 4)][(kk & 7) + ((kk >> 4) << 3)] = (bf16)h2;
            }
        }
        __syncthreads();
    }
}

// ---------------------------------------------------------------------------
// Segment-scan LSTM gate update (step k), all 8192 streams at once.
__global__ __launch_bounds__(256) void k_segpt(
    const float* __restrict__ U,   const float* __restrict__ pre,  // [8192][1024]
    const float* __restrict__ b1,  const float* __restrict__ b2,   // [1024]
    float* __restrict__ Hst, float* __restrict__ Cst,              // [8192][256]
    bf16* __restrict__ outk,                                       // [8192][256]
    int k, int dirRev)
{
    int n = blockIdx.x;
    int hh = threadIdx.x;
    int t = n >> 4, b = n & 15;
    int tk = dirRev ? (t - k) : (t + k);
    bool vx = dirRev ? (tk >= 0) : (tk < T_);
    const float* ur = U + (size_t)(vx ? (tk * B_ + b) : 0) * G4_;
    const float* pr = pre + (size_t)n * G4_;
    float gi = pr[hh]       + b1[hh]       + b2[hh]       + (vx ? ur[hh]       : 0.f);
    float gf = pr[256 + hh] + b1[256 + hh] + b2[256 + hh] + (vx ? ur[256 + hh] : 0.f);
    float gg = pr[512 + hh] + b1[512 + hh] + b2[512 + hh] + (vx ? ur[512 + hh] : 0.f);
    float go = pr[768 + hh] + b1[768 + hh] + b2[768 + hh] + (vx ? ur[768 + hh] : 0.f);
    size_t idx = (size_t)n * HH_ + hh;
    float c2 = sigf(gf) * Cst[idx] + sigf(gi) * tanhf(gg);
    float h2 = sigf(go) * tanhf(c2);
    Cst[idx] = c2;
    Hst[idx] = h2;
    outk[idx] = (bf16)h2;
}

// ---------------------------------------------------------------------------
// Final: gather segment features (masked) and FC to NTAG.
__global__ __launch_bounds__(128) void k_fc(
    const bf16* __restrict__ outF, const bf16* __restrict__ outR, // [8][8192][256]
    const float* __restrict__ Wfc, const float* __restrict__ bfc,
    const int* __restrict__ lens,
    float* __restrict__ outp)                                     // [B][T][P][NTAG]
{
    __shared__ float Ws[NTAG_ * 520];
    __shared__ float bs[NTAG_];
    __shared__ float feat[H_];
    int t = blockIdx.x, b = blockIdx.y;
    int tid = threadIdx.x;
    for (int i = tid; i < NTAG_ * H_; i += 128) Ws[(i / H_) * 520 + (i % H_)] = Wfc[i];
    if (tid < NTAG_) bs[tid] = bfc[tid];
    int lv = lens[b];
    __syncthreads();
    for (int p = 0; p < P_; ++p) {
        bool valid = (t >= p) && (t < lv);
#pragma unroll
        for (int q = 0; q < 4; ++q) {
            int j = tid * 4 + q;
            float v = 0.f;
            if (valid) {
                if (j < HH_)
                    v = (float)outF[((size_t)p * NR_ + (size_t)(t - p) * B_ + b) * HH_ + j];
                else
                    v = (float)outR[((size_t)p * NR_ + (size_t)t * B_ + b) * HH_ + (j - HH_)];
            }
            feat[j] = v;
        }
        __syncthreads();
        if (tid < NTAG_) {
            float s = bs[tid];
            for (int j = 0; j < H_; ++j) s += feat[j] * Ws[tid * 520 + j];
            outp[(((size_t)b * T_ + t) * P_ + p) * NTAG_ + tid] = s;
        }
        __syncthreads();
    }
}

// ---------------------------------------------------------------------------
static inline void gemm_launch(hipStream_t s, const float* A, int lda, const bf16* WT,
                               const float* b1, const float* b2,
                               float* C, int ldc, int M, int N, int K) {
    dim3 g(N / 128, M / 128);
    k_gemm<<<g, 256, 0, s>>>(A, lda, WT, b1, b2, C, ldc, M, N, K);
}
static inline void fill0(hipStream_t s, float* p, int n) {
    k_fill0<<<(n + 255) / 256, 256, 0, s>>>(p, n);
}

extern "C" void kernel_launch(void* const* d_in, const int* in_sizes, int n_in,
                              void* d_out, int out_size, void* d_ws, size_t ws_size,
                              hipStream_t stream)
{
    const float* x      = (const float*)d_in[0];
    const float* h0     = (const float*)d_in[1];
    const float* c0     = (const float*)d_in[2];
    const float* W_emb  = (const float*)d_in[3];
    const float* b_emb  = (const float*)d_in[4];
    const float* Wih    = (const float*)d_in[5];   // [4][1024][512]
    const float* Whh    = (const float*)d_in[6];   // [4][1024][256]
    const float* bih    = (const float*)d_in[7];   // [4][1024]
    const float* bhh    = (const float*)d_in[8];
    const float* Wih_sf = (const float*)d_in[9];
    const float* Whh_sf = (const float*)d_in[10];
    const float* bih_sf = (const float*)d_in[11];
    const float* bhh_sf = (const float*)d_in[12];
    const float* Wih_sr = (const float*)d_in[13];
    const float* Whh_sr = (const float*)d_in[14];
    const float* bih_sr = (const float*)d_in[15];
    const float* bhh_sr = (const float*)d_in[16];
    const float* Wfc    = (const float*)d_in[17];
    const float* bfc    = (const float*)d_in[18];
    const int*   lens   = (const int*)d_in[19];
    float* out = (float*)d_out;
    (void)in_sizes; (void)n_in; (void)out_size; (void)ws_size;

    // workspace layout
    char* cur = (char*)d_ws;
    auto take = [&](size_t bytes) { char* p = cur; cur += (bytes + 255) & ~(size_t)255; return p; };
    float* inpA  = (float*)take((size_t)NR_ * H_ * 4);
    float* inpB  = (float*)take((size_t)NR_ * H_ * 4);
    float* xWf   = (float*)take((size_t)NR_ * G4_ * 4);
    float* xWr   = (float*)take((size_t)NR_ * G4_ * 4);
    float* pre   = (float*)take((size_t)NR_ * G4_ * 4);
    float* Hseg  = (float*)take((size_t)NR_ * HH_ * 4);
    float* Cseg  = (float*)take((size_t)NR_ * HH_ * 4);
    bf16*  outF  = (bf16*)take((size_t)P_ * NR_ * HH_ * 2);
    bf16*  outR  = (bf16*)take((size_t)P_ * NR_ * HH_ * 2);
    bf16*  WTemb = (bf16*)take((size_t)CENC_ * EMB_ * 2);     // [128][512]
    bf16*  WTih  = (bf16*)take((size_t)4 * H_ * G4_ * 2);     // 4 x [512][1024]
    bf16*  WTsf  = (bf16*)take((size_t)H_ * G4_ * 2);         // [512][1024]
    bf16*  WTsr  = (bf16*)take((size_t)H_ * G4_ * 2);
    bf16*  WThf  = (bf16*)take((size_t)HH_ * G4_ * 2);        // [256][1024]
    bf16*  WThr  = (bf16*)take((size_t)HH_ * G4_ * 2);
    bf16*  WTenc = (bf16*)take((size_t)4 * HH_ * G4_ * 2);    // 4 x [256][1024]

    auto tconv = [&](const float* src, bf16* dst, int N, int K) {
        int tot = N * K;
        k_t<<<(tot + 255) / 256, 256, 0, stream>>>(src, dst, N, K);
    };

    // 0) pre-transpose all weights to bf16 [K][N]
    tconv(W_emb, WTemb, EMB_, CENC_);
    for (int i = 0; i < 4; ++i)
        tconv(Wih + (size_t)i * G4_ * H_, WTih + (size_t)i * H_ * G4_, G4_, H_);
    tconv(Wih_sf, WTsf, G4_, H_);
    tconv(Wih_sr, WTsr, G4_, H_);
    tconv(Whh_sf, WThf, G4_, HH_);
    tconv(Whh_sr, WThr, G4_, HH_);
    for (int i = 0; i < 4; ++i)
        tconv(Whh + (size_t)i * G4_ * HH_, WTenc + (size_t)i * HH_ * G4_, G4_, HH_);

    // 1) char embedding: inpA = x @ W_emb^T + b_emb
    gemm_launch(stream, x, CENC_, WTemb, b_emb, nullptr, inpA, H_, NR_, EMB_, CENC_);

    // 2) encoder layers
    const float* curIn = inpA;
    float* nxt = inpB;
    for (int l = 0; l < NL_; ++l) {
        gemm_launch(stream, curIn, H_, WTih + (size_t)(2 * l) * H_ * G4_,
                    bih + (2 * l) * G4_, bhh + (2 * l) * G4_, xWf, G4_, NR_, G4_, H_);
        gemm_launch(stream, curIn, H_, WTih + (size_t)(2 * l + 1) * H_ * G4_,
                    bih + (2 * l + 1) * G4_, bhh + (2 * l + 1) * G4_, xWr, G4_, NR_, G4_, H_);
        fill0(stream, nxt, NR_ * H_); // rev half rows >= len stay zero
        k_enc<<<2, 512, 0, stream>>>(xWf, xWr, WTenc, h0, c0, lens, nxt, l);
        const float* tmp = curIn; curIn = nxt; nxt = (float*)tmp;
    }
    const float* u = curIn; // encoder output [8192][512], rows t*B+b

    // 3) segment-scan input projections (shift-invariant, hoisted)
    gemm_launch(stream, u, H_, WTsf, nullptr, nullptr, xWf, G4_, NR_, G4_, H_);
    gemm_launch(stream, u, H_, WTsr, nullptr, nullptr, xWr, G4_, NR_, G4_, H_);

    // 4) segment scans: 8 steps each direction
    fill0(stream, Hseg, NR_ * HH_);
    fill0(stream, Cseg, NR_ * HH_);
    for (int k = 0; k < P_; ++k) {
        gemm_launch(stream, Hseg, HH_, WThf, nullptr, nullptr, pre, G4_, NR_, G4_, HH_);
        k_segpt<<<NR_, 256, 0, stream>>>(xWf, pre, bih_sf, bhh_sf, Hseg, Cseg,
                                         outF + (size_t)k * NR_ * HH_, k, 0);
    }
    fill0(stream, Hseg, NR_ * HH_);
    fill0(stream, Cseg, NR_ * HH_);
    for (int k = 0; k < P_; ++k) {
        gemm_launch(stream, Hseg, HH_, WThr, nullptr, nullptr, pre, G4_, NR_, G4_, HH_);
        k_segpt<<<NR_, 256, 0, stream>>>(xWr, pre, bih_sr, bhh_sr, Hseg, Cseg,
                                         outR + (size_t)k * NR_ * HH_, k, 1);
    }

    // 5) gather + mask + FC
    {
        dim3 g(T_, B_);
        k_fc<<<g, 128, 0, stream>>>(outF, outR, Wfc, bfc, lens, out);
    }
}
